// SAGE_29291676959272
// MI455X (gfx1250) — matrix operand order repaired and checked by reading, not compiled, not run on hardware
//
#include <hip/hip_runtime.h>

// SAGE pool (2 layers, D=64) for gfx1250 — dense math on v_wmma_f32_16x16x32_f16.
//
// Pass 0: pack 6 weight matrices (f32, row-major) into per-lane f16 B-fragment
//         layout [nt][ks][lane][e] so hot kernels do one 32B vector load per
//         fragment (no cvt VALU, fully coalesced).
// Pass 1: per-dst fused gather+transform+relu+segmax (x f32 -> h_neigh1 f16)
// Pass 2: combine tanh(x[:N0]*Ws1^T + neigh1*Wn1^T + b1) -> h f16
// Pass 3: per-dst fused gather+transform+relu+segmax (h f16 -> h_neigh2 f16)
// Pass 4: combine h[:N1]*Ws2^T + neigh2*Wn2^T + b2 -> d_out f32
//
// Pool kernels are persistent grid-stride waves: B fragments (64 VGPRs) loaded
// once, then only the 256B gathered rows move per iteration (HBM-bound shape:
// E0 * 256B ~= 570MB => ~25us floor at 23.3 TB/s; WMMA work is far below the
// f16 matrix-pipe ceiling, so it stays off the critical path).

typedef __attribute__((ext_vector_type(16))) _Float16 v16h;
typedef __attribute__((ext_vector_type(8)))  float    v8f;

#define D 64
#define FANOUT 16
#define NMAT 6                   // Wp1, Wn1, Ws1, Wp2, Wn2, Ws2
#define WPACK (4 * 2 * 32 * 16)  // 4096 f16 per packed matrix

// rows 0-7 <-> 8-15 of the D tile live in opposite lane halves:
// ds_swizzle SWAPX16 (offset[14:10]=xor=0x10, and=0x1f) swaps lane L <-> L^16.
__device__ __forceinline__ float swap16(float v) {
  return __int_as_float(__builtin_amdgcn_ds_swizzle(__float_as_int(v), 0x401F));
}

// packed B fragment: one contiguous 32B load per lane
__device__ __forceinline__ v16h load_bp(const _Float16* __restrict__ P,
                                        int lane, int nt, int ks) {
  return *(const v16h*)(P + ((size_t)((((nt << 1) | ks) << 5) | lane) << 4));
}

// 16-bit A 16x32 layout: lane L holds row m=L&15; elems 0..7 -> K=8*(L>>4)+e,
// elems 8..15 -> K=16+8*(L>>4)+(e-8).
__device__ __forceinline__ v16h load_a_f32(const float* __restrict__ row,
                                           int lane, int ks) {
  const int base = ks * 32 + ((lane >> 4) & 1) * 8;
  v16h a;
#pragma unroll
  for (int e = 0; e < 8; ++e) {
    a[e]     = (_Float16)row[base + e];
    a[e + 8] = (_Float16)row[base + 16 + e];
  }
  return a;
}

__device__ __forceinline__ v16h load_a_f16(const _Float16* __restrict__ row,
                                           int lane, int ks) {
  const int base = ks * 32 + ((lane >> 4) & 1) * 8;
  v16h a;
#pragma unroll
  for (int e = 0; e < 8; ++e) {
    a[e]     = row[base + e];
    a[e + 8] = row[base + 16 + e];
  }
  return a;
}

// ---- Pass 0: pack weights into B-fragment layout ---------------------------
// B(k,n) = W[nt*16+n][ks*32+k]; fragment elem e of lane L is
// W[nt*16 + (L&15)][ks*32 + 8*(L>>4) + e + (e>=8 ? 8 : 0)].
__global__ __launch_bounds__(256) void pack_weights_kernel(
    const float* __restrict__ W0, const float* __restrict__ W1,
    const float* __restrict__ W2, const float* __restrict__ W3,
    const float* __restrict__ W4, const float* __restrict__ W5,
    _Float16* __restrict__ out) {
  const int tid = blockIdx.x * blockDim.x + threadIdx.x;
  if (tid >= NMAT * WPACK) return;
  const int mat = tid >> 12;
  const int idx = tid & (WPACK - 1);
  const float* W = (mat == 0) ? W0 : (mat == 1) ? W1 : (mat == 2) ? W2
                 : (mat == 3) ? W3 : (mat == 4) ? W4 : W5;
  const int e  = idx & 15;
  const int L  = (idx >> 4) & 31;
  const int ks = (idx >> 9) & 1;
  const int nt = (idx >> 10) & 3;
  const int col = ks * 32 + ((L >> 4) & 1) * 8 + e + ((e >> 3) << 3);
  out[tid] = (_Float16)W[(size_t)(nt * 16 + (L & 15)) * D + col];
}

// ---- Pass 1 / 3: fused  max_e relu(x[src]*Wp^T + bp)  per destination ------
// One wave per destination per loop trip: the dst's 16 gathered rows form one
// A tile; 8 WMMA; relu+max reduce across the 16 rows (in-lane max3 tree + one
// cross-half ds_swizzle). EXEC stays all-1s everywhere (WMMA requirement).
template <bool SRC_F16>
__global__ __launch_bounds__(256) void pool_neigh_kernel(
    const void* __restrict__ xsrc, const int* __restrict__ src,
    const _Float16* __restrict__ Wp_pk, const float* __restrict__ bp,
    _Float16* __restrict__ out_neigh, int n_dst) {
  const int lane   = threadIdx.x & 31;
  const int wave0  = (int)((blockIdx.x * blockDim.x + threadIdx.x) >> 5);
  const int nwaves = (int)((gridDim.x * blockDim.x) >> 5);

  // resident weights + bias
  v16h b[4][2];
#pragma unroll
  for (int nt = 0; nt < 4; ++nt)
#pragma unroll
    for (int ks = 0; ks < 2; ++ks) b[nt][ks] = load_bp(Wp_pk, lane, nt, ks);
  float bias[4];
#pragma unroll
  for (int nt = 0; nt < 4; ++nt) bias[nt] = bp[nt * 16 + (lane & 15)];

  const int m = lane & 15;  // edge slot -> A row
  for (int d = wave0; d < n_dst; d += nwaves) {
    const int r = src[(size_t)d * FANOUT + m];
    v16h a0, a1;
    if (SRC_F16) {
      const _Float16* row = (const _Float16*)xsrc + (size_t)r * D;
      a0 = load_a_f16(row, lane, 0);
      a1 = load_a_f16(row, lane, 1);
    } else {
      const float* row = (const float*)xsrc + (size_t)r * D;
      a0 = load_a_f32(row, lane, 0);
      a1 = load_a_f32(row, lane, 1);
    }

    float vmax[4];
#pragma unroll
    for (int nt = 0; nt < 4; ++nt) {
      v8f acc;
#pragma unroll
      for (int e = 0; e < 8; ++e) acc[e] = bias[nt];
      acc = __builtin_amdgcn_wmma_f32_16x16x32_f16(false, a0, false, b[nt][0],
                                                   (short)0, acc, false, false);
      acc = __builtin_amdgcn_wmma_f32_16x16x32_f16(false, a1, false, b[nt][1],
                                                   (short)0, acc, false, false);
      float vm = 0.0f;  // relu floor folds into segment_max
#pragma unroll
      for (int e = 0; e < 8; ++e) vm = fmaxf(vm, acc[e]);
      vmax[nt] = fmaxf(vm, swap16(vm));  // both halves now hold column max
    }
    // all-lane coalesced store: feature `lane` and `32+lane`
    const int hi = (lane >> 4) & 1;
    const float v0 = hi ? vmax[1] : vmax[0];
    const float v1 = hi ? vmax[3] : vmax[2];
    out_neigh[(size_t)d * D + lane]      = (_Float16)v0;
    out_neigh[(size_t)d * D + 32 + lane] = (_Float16)v1;
  }
}

// ---- Pass 2 / 4: out = act(self*Ws^T + neigh*Wn^T + b) ---------------------
// One wave per 16-row tile; 16 WMMA per tile (4 N-tiles x (self+neigh) x 2 K).
template <bool SELF_F16, bool TANH, bool OUT_F16>
__global__ __launch_bounds__(256) void combine_kernel(
    const void* __restrict__ hself, const _Float16* __restrict__ hneigh,
    const _Float16* __restrict__ Ws_pk, const _Float16* __restrict__ Wn_pk,
    const float* __restrict__ bias, void* __restrict__ out, int n_tiles) {
  const int lane   = threadIdx.x & 31;
  const int wave0  = (int)((blockIdx.x * blockDim.x + threadIdx.x) >> 5);
  const int nwaves = (int)((gridDim.x * blockDim.x) >> 5);

  v16h bs[4][2], bn[4][2];
#pragma unroll
  for (int nt = 0; nt < 4; ++nt)
#pragma unroll
    for (int ks = 0; ks < 2; ++ks) {
      bs[nt][ks] = load_bp(Ws_pk, lane, nt, ks);
      bn[nt][ks] = load_bp(Wn_pk, lane, nt, ks);
    }
  float bv[4];
#pragma unroll
  for (int nt = 0; nt < 4; ++nt) bv[nt] = bias[nt * 16 + (lane & 15)];

  const int m = lane & 15;
  for (int t = wave0; t < n_tiles; t += nwaves) {
    const int d0 = t * 16;
    v16h as0, as1, an0, an1;
    if (SELF_F16) {
      const _Float16* row = (const _Float16*)hself + (size_t)(d0 + m) * D;
      as0 = load_a_f16(row, lane, 0);
      as1 = load_a_f16(row, lane, 1);
    } else {
      const float* row = (const float*)hself + (size_t)(d0 + m) * D;
      as0 = load_a_f32(row, lane, 0);
      as1 = load_a_f32(row, lane, 1);
    }
    {
      const _Float16* row = hneigh + (size_t)(d0 + m) * D;
      an0 = load_a_f16(row, lane, 0);
      an1 = load_a_f16(row, lane, 1);
    }

#pragma unroll
    for (int nt = 0; nt < 4; ++nt) {
      v8f acc;
#pragma unroll
      for (int e = 0; e < 8; ++e) acc[e] = bv[nt];
      acc = __builtin_amdgcn_wmma_f32_16x16x32_f16(false, as0, false, bs[nt][0],
                                                   (short)0, acc, false, false);
      acc = __builtin_amdgcn_wmma_f32_16x16x32_f16(false, as1, false, bs[nt][1],
                                                   (short)0, acc, false, false);
      acc = __builtin_amdgcn_wmma_f32_16x16x32_f16(false, an0, false, bn[nt][0],
                                                   (short)0, acc, false, false);
      acc = __builtin_amdgcn_wmma_f32_16x16x32_f16(false, an1, false, bn[nt][1],
                                                   (short)0, acc, false, false);
      // D layout: lane holds column n = nt*16+(lane&15), rows 8*(lane>>4)+e
      const int n = nt * 16 + (lane & 15);
      const int rbase = d0 + ((lane >> 4) & 1) * 8;
#pragma unroll
      for (int e = 0; e < 8; ++e) {
        float v = acc[e];
        if (TANH) v = tanhf(v);
        if (OUT_F16)
          ((_Float16*)out)[(size_t)(rbase + e) * D + n] = (_Float16)v;
        else
          ((float*)out)[(size_t)(rbase + e) * D + n] = v;
      }
    }
  }
}

// ---- host-side launch ------------------------------------------------------
extern "C" void kernel_launch(void* const* d_in, const int* in_sizes, int n_in,
                              void* d_out, int out_size, void* d_ws,
                              size_t ws_size, hipStream_t stream) {
  const float* x   = (const float*)d_in[0];
  const int* src0  = (const int*)d_in[1];
  const int* src1  = (const int*)d_in[3];
  const float* Wp1 = (const float*)d_in[7];
  const float* bp1 = (const float*)d_in[8];
  const float* Wn1 = (const float*)d_in[9];
  const float* Ws1 = (const float*)d_in[10];
  const float* b1  = (const float*)d_in[11];
  const float* Wp2 = (const float*)d_in[12];
  const float* bp2 = (const float*)d_in[13];
  const float* Wn2 = (const float*)d_in[14];
  const float* Ws2 = (const float*)d_in[15];
  const float* b2  = (const float*)d_in[16];

  const int E0 = in_sizes[1];
  const int E1 = in_sizes[3];
  const int n_dst0 = E0 / FANOUT;  // 139264
  const int n_dst1 = E1 / FANOUT;  // 8192

  // workspace layout
  char* ws = (char*)d_ws;
  _Float16* wpk = (_Float16*)ws;  // 6 * 4096 f16 = 48KB
  _Float16* Wp1p = wpk + 0 * WPACK;
  _Float16* Wn1p = wpk + 1 * WPACK;
  _Float16* Ws1p = wpk + 2 * WPACK;
  _Float16* Wp2p = wpk + 3 * WPACK;
  _Float16* Wn2p = wpk + 4 * WPACK;
  _Float16* Ws2p = wpk + 5 * WPACK;
  char* p = ws + (size_t)NMAT * WPACK * sizeof(_Float16);
  _Float16* hneigh1 = (_Float16*)p;                            // N0*64 f16
  _Float16* h       = (_Float16*)(p + (size_t)n_dst0 * D * 2); // N0*64 f16
  _Float16* hneigh2 = (_Float16*)(p + (size_t)n_dst0 * D * 4); // N1*64 f16

  const int BLK = 256;  // 8 waves/block

  // Pass 0: pack weights (runs once per launch, ~24K threads)
  pack_weights_kernel<<<(NMAT * WPACK + BLK - 1) / BLK, BLK, 0, stream>>>(
      Wp1, Wn1, Ws1, Wp2, Wn2, Ws2, wpk);

  auto grid_for = [BLK](int waves, int cap) {
    int g = (waves * 32 + BLK - 1) / BLK;
    return g < cap ? g : cap;
  };

  // Pass 1: layer-1 fused pool (persistent waves; dominant, gather-bound pass)
  pool_neigh_kernel<false><<<grid_for(n_dst0, 2048), BLK, 0, stream>>>(
      x, src0, Wp1p, bp1, hneigh1, n_dst0);
  // Pass 2: layer-1 combine + tanh -> h (f16)
  combine_kernel<false, true, true><<<grid_for(n_dst0 / 16, 512), BLK, 0,
                                      stream>>>(x, hneigh1, Ws1p, Wn1p, b1, h,
                                                n_dst0 / 16);
  // Pass 3: layer-2 fused pool
  pool_neigh_kernel<true><<<grid_for(n_dst1, 256), BLK, 0, stream>>>(
      h, src1, Wp2p, bp2, hneigh2, n_dst1);
  // Pass 4: layer-2 combine -> d_out (f32)
  combine_kernel<true, false, false><<<grid_for(n_dst1 / 16, 64), BLK, 0,
                                       stream>>>(h, hneigh2, Ws2p, Wn2p, b2,
                                                 d_out, n_dst1 / 16);
}
